// FramAnglNet_76020921139676
// MI455X (gfx1250) — matrix operand order repaired and checked by reading, not compiled, not run on hardware
//
#include <hip/hip_runtime.h>

// ---------------- types ----------------
typedef __attribute__((ext_vector_type(16))) __bf16        v16bf;
typedef __attribute__((ext_vector_type(8)))  float         v8f;
typedef __attribute__((ext_vector_type(4)))  unsigned int  u32x4;

#define DS   384
#define DH   128
#define KOUT 14
#define WAVES_PER_BLOCK 4
#define ROWS_PER_WAVE   16
#define ROWS_PER_BLOCK  (WAVES_PER_BLOCK * ROWS_PER_WAVE)

// bf16 workspace layout (element offsets)
#define NWS   (128 * 384)
#define NW    (128 * 128)
#define NW3   (16 * 128)
#define OFF_WS   0
#define OFF_W11  (NWS)
#define OFF_W12  (NWS + 1 * NW)
#define OFF_W21  (NWS + 2 * NW)
#define OFF_W22  (NWS + 3 * NW)
#define OFF_W3   (NWS + 4 * NW)
#define WTOTAL   (NWS + 4 * NW + NW3)

// LDS layout (element offsets into 32768-short / 64KB arena)
//  region A [0, 8192): per-wave 16x128 staging tiles (2048 each);
//                      aliased as 4096-short Ws K-slice during Ws GEMM
//  region B [8192, 32768): per-wave 16x384 x tiles (6144 each);
//                      aliased as 16384-short weight buffer for resblocks
#define LDS_SHORTS 32768
#define HST_OFF(w) ((w) * 2048)
#define WSLICE_OFF 0
#define XS_OFF(w)  (8192 + (w) * 6144)
#define WBUF_OFF   8192

// ---------------- helpers ----------------
static __device__ __forceinline__ unsigned short f2bf(float f) {
  unsigned u = __float_as_uint(f);
  u += 0x7FFFu + ((u >> 16) & 1u);          // round-to-nearest-even
  return (unsigned short)(u >> 16);
}

static __device__ __forceinline__ v16bf frag2(const unsigned short* p0,
                                              const unsigned short* p1) {
  union { u32x4 u[2]; v16bf v; } c;
  c.u[0] = *(const u32x4*)p0;               // 16B -> ds/global load_b128
  c.u[1] = *(const u32x4*)p1;
  return c.v;
}

// A-matrix frag (16-bit, 16x32): lanes 0-15 row m hold K kt..kt+15 contiguous,
// lanes 16-31 same rows hold K kt+16..kt+31 (matches loadB).
static __device__ __forceinline__ v16bf loadA(const unsigned short* base, int lda,
                                              int kt, int lane) {
  const int row = lane & 15;
  const int kb  = kt + ((lane >> 4) << 4);
  const unsigned short* p = base + row * lda + kb;
  return frag2(p, p + 8);
}

// B-matrix frag (16-bit, 32x16): lanes 0-15 col n hold K kt..kt+15 contiguous,
// lanes 16-31 hold K kt+16..kt+31.
static __device__ __forceinline__ v16bf loadB(const unsigned short* base, int ldb,
                                              int n, int kt, int lane) {
  const int kb = kt + ((lane >> 4) << 4);
  const unsigned short* p = base + n * ldb + kb;
  return frag2(p, p + 8);
}

template <int NT, int K>
static __device__ __forceinline__ void gemm_acc(v8f* acc, const unsigned short* A,
                                                int lda, const unsigned short* W,
                                                int ldw, int lane) {
#pragma unroll
  for (int kt = 0; kt < K; kt += 32) {
    v16bf af = loadA(A, lda, kt, lane);
#pragma unroll
    for (int ct = 0; ct < NT; ct++) {
      const int n = ct * 16 + (lane & 15);
      v16bf wf = loadB(W, ldw, n, kt, lane);
      acc[ct] = __builtin_amdgcn_wmma_f32_16x16x32_bf16(
          false, af, false, wf, (short)0, acc[ct], false, false);
    }
  }
}

// D layout (32-bit C/D 16x16): element j -> row j + 8*(lane>=16), col lane&15
template <bool RELU>
static __device__ __forceinline__ void stage16(unsigned short* hst, const v8f* acc,
                                               int lane) {
  const int col = lane & 15;
  const int rb  = (lane >> 4) << 3;
#pragma unroll
  for (int ct = 0; ct < 8; ct++)
#pragma unroll
    for (int j = 0; j < 8; j++) {
      float e = acc[ct][j];
      if (RELU) e = fmaxf(e, 0.0f);
      hst[(rb + j) * DH + ct * 16 + col] = f2bf(e);
    }
}

static __device__ __forceinline__ void set_bias(v8f* acc, const float* b, int lane) {
  const int col = lane & 15;
#pragma unroll
  for (int ct = 0; ct < 8; ct++) {
    const float bv = b[ct * 16 + col];
#pragma unroll
    for (int j = 0; j < 8; j++) acc[ct][j] = bv;
  }
}
static __device__ __forceinline__ void add_bias(v8f* acc, const float* b, int lane) {
  const int col = lane & 15;
#pragma unroll
  for (int ct = 0; ct < 8; ct++) {
    const float bv = b[ct * 16 + col];
#pragma unroll
    for (int j = 0; j < 8; j++) acc[ct][j] += bv;
  }
}

// cooperative load of a row-major [rows x 128] bf16 weight matrix into LDS
static __device__ __forceinline__ void coop_load_w(unsigned short* dst,
                                                   const unsigned short* src,
                                                   int rows) {
  if ((int)threadIdx.x < rows) {
    const u32x4* s = (const u32x4*)(src + threadIdx.x * DH);
    u32x4* d = (u32x4*)(dst + threadIdx.x * DH);
#pragma unroll
    for (int i = 0; i < 16; i++) d[i] = s[i];
  }
}

// quat (b,c,d), implied a=1, normalized rotation
static __device__ __forceinline__ void q2r(float b, float c, float d, float* R) {
  const float s  = 1.0f + b * b + c * c + d * d;
  const float is = 1.0f / s;
  R[0] = (1.0f + b * b - c * c - d * d) * is;
  R[1] = 2.0f * (b * c - d) * is;
  R[2] = 2.0f * (b * d + c) * is;
  R[3] = 2.0f * (b * c + d) * is;
  R[4] = (1.0f - b * b + c * c - d * d) * is;
  R[5] = 2.0f * (c * d - b) * is;
  R[6] = 2.0f * (b * d - c) * is;
  R[7] = 2.0f * (c * d + b) * is;
  R[8] = (1.0f - b * b - c * c + d * d) * is;
}

// ---------------- weight convert (f32 -> bf16, W3 padded to 16 rows) ----------
__global__ void convert_weights(const float* __restrict__ Ws,
                                const float* __restrict__ W11,
                                const float* __restrict__ W12,
                                const float* __restrict__ W21,
                                const float* __restrict__ W22,
                                const float* __restrict__ W3,
                                unsigned short* __restrict__ out) {
  int i = blockIdx.x * blockDim.x + threadIdx.x;
  if (i >= WTOTAL) return;
  float v;
  if (i < NWS) {
    v = Ws[i];
  } else {
    int j = i - NWS;
    if (j < NW)            v = W11[j];
    else if ((j -= NW) < NW) v = W12[j];
    else if ((j -= NW) < NW) v = W21[j];
    else if ((j -= NW) < NW) v = W22[j];
    else { j -= NW; v = (j < KOUT * DH) ? W3[j] : 0.0f; }
  }
  out[i] = f2bf(v);
}

// ---------------- fused kernel ----------------
__global__ __launch_bounds__(128) void fram_fused(
    const float* __restrict__ sfea, const float* __restrict__ quat,
    const float* __restrict__ trsl, const float* __restrict__ ln_g,
    const float* __restrict__ ln_b, const float* __restrict__ Wq,
    const float* __restrict__ bq, const float* __restrict__ Wt,
    const float* __restrict__ bt, const unsigned short* __restrict__ WsB,
    const float* __restrict__ bs, const unsigned short* __restrict__ W11B,
    const float* __restrict__ b11, const unsigned short* __restrict__ W12B,
    const float* __restrict__ b12, const unsigned short* __restrict__ W21B,
    const float* __restrict__ b21, const unsigned short* __restrict__ W22B,
    const float* __restrict__ b22, const unsigned short* __restrict__ W3B,
    const float* __restrict__ b3, float* __restrict__ oq, float* __restrict__ ot,
    float* __restrict__ oa, int NL) {
  __shared__ __align__(16) unsigned short smem[LDS_SHORTS];  // 64 KB

  const int lane = threadIdx.x & 31;
  const int wave = threadIdx.x >> 5;
  unsigned short* xs   = smem + XS_OFF(wave);   // 16 x 384 bf16 (wave private)
  unsigned short* hst  = smem + HST_OFF(wave);  // 16 x 128 staging (wave private)
  unsigned short* wbuf = smem + WBUF_OFF;       // 128 x 128 shared weight buffer
  const int rowBase = blockIdx.x * ROWS_PER_BLOCK + wave * ROWS_PER_WAVE;

  // ---- phase 1: LayerNorm + quat/trsl heads, one wave per 16 rows ----
  float g[12], bb[12], wqr[3][12], wtr[3][12];
#pragma unroll
  for (int i = 0; i < 12; i++) {
    const int idx = lane + 32 * i;
    g[i]  = ln_g[idx];
    bb[i] = ln_b[idx];
#pragma unroll
    for (int j = 0; j < 3; j++) {
      wqr[j][i] = Wq[j * DS + idx];
      wtr[j][i] = Wt[j * DS + idx];
    }
  }

  for (int r = 0; r < ROWS_PER_WAVE; r++) {
    const int row  = rowBase + r;
    const int rowc = (row < NL) ? row : (NL - 1);
    const float* xin = sfea + (size_t)rowc * DS;

    float v[12], s = 0.0f, s2 = 0.0f;
#pragma unroll
    for (int i = 0; i < 12; i++) {
      v[i] = xin[lane + 32 * i];
      s += v[i];
      s2 += v[i] * v[i];
    }
    for (int o = 16; o; o >>= 1) {
      s  += __shfl_xor(s, o, 32);
      s2 += __shfl_xor(s2, o, 32);
    }
    const float m    = s * (1.0f / DS);
    const float var  = s2 * (1.0f / DS) - m * m;
    const float rstd = rsqrtf(var + 1e-5f);

    float a0 = 0, a1 = 0, a2 = 0, t0 = 0, t1 = 0, t2 = 0;
#pragma unroll
    for (int i = 0; i < 12; i++) {
      const float xi = (v[i] - m) * rstd * g[i] + bb[i];
      xs[r * DS + lane + 32 * i] = f2bf(xi);
      a0 += xi * wqr[0][i]; a1 += xi * wqr[1][i]; a2 += xi * wqr[2][i];
      t0 += xi * wtr[0][i]; t1 += xi * wtr[1][i]; t2 += xi * wtr[2][i];
    }
    for (int o = 16; o; o >>= 1) {
      a0 += __shfl_xor(a0, o, 32); a1 += __shfl_xor(a1, o, 32);
      a2 += __shfl_xor(a2, o, 32); t0 += __shfl_xor(t0, o, 32);
      t1 += __shfl_xor(t1, o, 32); t2 += __shfl_xor(t2, o, 32);
    }
    if (lane == 0 && row < NL) {
      const float qb = a0 + bq[0], qc = a1 + bq[1], qd = a2 + bq[2];
      const float u0 = t0 + bt[0], u1 = t1 + bt[1], u2 = t2 + bt[2];
      float Ro[9], Ru[9], Rn[9];
      q2r(quat[(size_t)row * 3 + 0], quat[(size_t)row * 3 + 1],
          quat[(size_t)row * 3 + 2], Ro);
      q2r(qb, qc, qd, Ru);
#pragma unroll
      for (int i = 0; i < 3; i++)
#pragma unroll
        for (int j = 0; j < 3; j++)
          Rn[i * 3 + j] = Ro[i * 3 + 0] * Ru[0 * 3 + j] +
                          Ro[i * 3 + 1] * Ru[1 * 3 + j] +
                          Ro[i * 3 + 2] * Ru[2 * 3 + j];
      const float inv = 1.0f / (1.0f + Rn[0] + Rn[4] + Rn[8]);
      oq[(size_t)row * 3 + 0] = (Rn[7] - Rn[5]) * inv;
      oq[(size_t)row * 3 + 1] = (Rn[2] - Rn[6]) * inv;
      oq[(size_t)row * 3 + 2] = (Rn[3] - Rn[1]) * inv;
#pragma unroll
      for (int i = 0; i < 3; i++)
        ot[(size_t)row * 3 + i] = trsl[(size_t)row * 3 + i] +
                                  Ro[i * 3 + 0] * u0 + Ro[i * 3 + 1] * u1 +
                                  Ro[i * 3 + 2] * u2;
    }
  }

  // ---- phase 2: h = x @ Ws^T + bs, Ws streamed through LDS in 32-K slices ----
  v8f h[8];
  set_bias(h, bs, lane);
  const int row16 = lane & 15;
  for (int kt = 0; kt < DS; kt += 32) {
    __syncthreads();
    {  // cooperative slice load: thread t -> feature row t, 32 K values (64B)
      const u32x4* s = (const u32x4*)(WsB + threadIdx.x * DS + kt);
      u32x4* d = (u32x4*)(smem + WSLICE_OFF + threadIdx.x * 32);
#pragma unroll
      for (int i = 0; i < 4; i++) d[i] = s[i];
    }
    __syncthreads();
    v16bf af = loadA(xs, DS, kt, lane);
#pragma unroll
    for (int ct = 0; ct < 8; ct++) {
      const unsigned short* p =
          smem + WSLICE_OFF + (ct * 16 + row16) * 32 + ((lane >> 4) << 4);
      v16bf wf = frag2(p, p + 8);
      h[ct] = __builtin_amdgcn_wmma_f32_16x16x32_bf16(false, af, false, wf,
                                                      (short)0, h[ct], false, false);
    }
  }

  // ---- residual block 1: h += relu(h@W11^T+b11) @ W12^T + b12 ----
  v8f t[8];
  __syncthreads();                      // all waves done with Ws GEMM / xs
  coop_load_w(wbuf, W11B, DH);
  stage16<false>(hst, h, lane);
  __syncthreads();
  set_bias(t, b11, lane);
  gemm_acc<8, DH>(t, hst, DH, wbuf, DH, lane);

  __syncthreads();                      // all waves done reading W11
  coop_load_w(wbuf, W12B, DH);
  stage16<true>(hst, t, lane);          // stages relu(t)
  __syncthreads();
  add_bias(h, b12, lane);
  gemm_acc<8, DH>(h, hst, DH, wbuf, DH, lane);

  // ---- residual block 2 ----
  __syncthreads();
  coop_load_w(wbuf, W21B, DH);
  stage16<false>(hst, h, lane);
  __syncthreads();
  set_bias(t, b21, lane);
  gemm_acc<8, DH>(t, hst, DH, wbuf, DH, lane);

  __syncthreads();
  coop_load_w(wbuf, W22B, DH);
  stage16<true>(hst, t, lane);
  __syncthreads();
  add_bias(h, b22, lane);
  gemm_acc<8, DH>(h, hst, DH, wbuf, DH, lane);

  // ---- angle head: angl = relu(h) @ W3^T + b3   (16 x 14, padded to 16) ----
  __syncthreads();
  coop_load_w(wbuf, W3B, 16);
  stage16<true>(hst, h, lane);
  __syncthreads();
  v8f a3;
  const int col = lane & 15;
  {
    const float bv = (col < KOUT) ? b3[col] : 0.0f;
#pragma unroll
    for (int j = 0; j < 8; j++) a3[j] = bv;
  }
  gemm_acc<1, DH>(&a3, hst, DH, wbuf, DH, lane);

  const int rb = (lane >> 4) << 3;
  if (col < KOUT) {
#pragma unroll
    for (int j = 0; j < 8; j++) {
      const int rg = rowBase + rb + j;
      if (rg < NL) oa[(size_t)rg * KOUT + col] = a3[j];
    }
  }
}

// ---------------- host launch ----------------
extern "C" void kernel_launch(void* const* d_in, const int* in_sizes, int n_in,
                              void* d_out, int out_size, void* d_ws, size_t ws_size,
                              hipStream_t stream) {
  const float* sfea = (const float*)d_in[0];
  // d_in[1] (sfea_tns_init) unused in v2 path
  const float* quat = (const float*)d_in[2];
  const float* trsl = (const float*)d_in[3];
  const float* ln_g = (const float*)d_in[4];
  const float* ln_b = (const float*)d_in[5];
  const float* Wq = (const float*)d_in[6];
  const float* bq = (const float*)d_in[7];
  const float* Wt = (const float*)d_in[8];
  const float* bt = (const float*)d_in[9];
  const float* Ws = (const float*)d_in[10];
  const float* bs = (const float*)d_in[11];
  const float* W11 = (const float*)d_in[12];
  const float* b11 = (const float*)d_in[13];
  const float* W12 = (const float*)d_in[14];
  const float* b12 = (const float*)d_in[15];
  const float* W21 = (const float*)d_in[16];
  const float* b21 = (const float*)d_in[17];
  const float* W22 = (const float*)d_in[18];
  const float* b22 = (const float*)d_in[19];
  const float* W3 = (const float*)d_in[20];
  const float* b3 = (const float*)d_in[21];

  unsigned short* wb = (unsigned short*)d_ws;
  const unsigned short* WsB  = wb + OFF_WS;
  const unsigned short* W11B = wb + OFF_W11;
  const unsigned short* W12B = wb + OFF_W12;
  const unsigned short* W21B = wb + OFF_W21;
  const unsigned short* W22B = wb + OFF_W22;
  const unsigned short* W3B  = wb + OFF_W3;

  const int NL = in_sizes[2] / 3;  // N*L
  float* oq = (float*)d_out;
  float* ot = oq + (size_t)NL * 3;
  float* oa = ot + (size_t)NL * 3;

  convert_weights<<<(WTOTAL + 255) / 256, 256, 0, stream>>>(Ws, W11, W12, W21, W22,
                                                            W3, wb);

  const int grid = (NL + ROWS_PER_BLOCK - 1) / ROWS_PER_BLOCK;
  fram_fused<<<grid, 128, 0, stream>>>(sfea, quat, trsl, ln_g, ln_b, Wq, bq, Wt, bt,
                                       WsB, bs, W11B, b11, W12B, b12, W21B, b21,
                                       W22B, b22, W3B, b3, oq, ot, oa, NL);
}
